// AttentionAssignmentNetwork_8340826489005
// MI455X (gfx1250) — matrix operand order repaired and checked by reading, not compiled, not run on hardware
//
#include <hip/hip_runtime.h>
#include <hip/hip_bf16.h>

#define NN 16384
#define HH 256
#define EE 4096

typedef __attribute__((ext_vector_type(16))) __bf16 v16bf;
typedef __attribute__((ext_vector_type(8)))  __bf16 v8bf;
typedef __attribute__((ext_vector_type(8)))  float  v8f;
typedef __attribute__((ext_vector_type(4)))  float  v4f;
typedef __attribute__((ext_vector_type(4)))  int    v4i;

// ---- CDNA5 async global->LDS copy (ASYNCcnt path), with portable fallback --
#if defined(__has_builtin) && __has_builtin(__builtin_amdgcn_global_load_async_to_lds_b128)
#define HAVE_ASYNC_LDS 1
typedef __attribute__((address_space(1))) v4i* gv4i_p;   // global int4*
typedef __attribute__((address_space(3))) v4i* lv4i_p;   // LDS int4*
// copies 16 bytes per lane, global -> LDS, no VGPR round-trip
#define ASYNC_CP16(g, l) \
    __builtin_amdgcn_global_load_async_to_lds_b128((gv4i_p)(g), (lv4i_p)(l), 0, 0)
#if __has_builtin(__builtin_amdgcn_s_wait_asynccnt)
#define ASYNC_WAIT() __builtin_amdgcn_s_wait_asynccnt(0)
#else
#define ASYNC_WAIT() asm volatile("s_wait_asynccnt 0x0" ::: "memory")
#endif
#else
#define HAVE_ASYNC_LDS 0
#define ASYNC_CP16(g, l) (*(v8bf*)(l) = *(const v8bf*)(g))
#define ASYNC_WAIT() ((void)0)
#endif

// ---- fragment loaders (WMMA 16x16x32 bf16, contiguous k = 16h+s in A and B) -------------------
// A 16x32: lane<16 holds row M=lane, K={0..15}; lane>=16 K={16..31}
// caller passes base = k0 + (lane>>4)*16
static __device__ __forceinline__ v16bf ldA_f32(const float* __restrict__ row, int base) {
    v4f x0 = *(const v4f*)(row + base);
    v4f x1 = *(const v4f*)(row + base + 4);
    v4f x2 = *(const v4f*)(row + base + 8);
    v4f x3 = *(const v4f*)(row + base + 12);
    v16bf r;
    r[0]=(__bf16)x0[0]; r[1]=(__bf16)x0[1]; r[2]=(__bf16)x0[2]; r[3]=(__bf16)x0[3];
    r[4]=(__bf16)x1[0]; r[5]=(__bf16)x1[1]; r[6]=(__bf16)x1[2]; r[7]=(__bf16)x1[3];
    r[8]=(__bf16)x2[0]; r[9]=(__bf16)x2[1]; r[10]=(__bf16)x2[2]; r[11]=(__bf16)x2[3];
    r[12]=(__bf16)x3[0]; r[13]=(__bf16)x3[1]; r[14]=(__bf16)x3[2]; r[15]=(__bf16)x3[3];
    return r;
}
static __device__ __forceinline__ v16bf ldA_bf16(const __bf16* __restrict__ row, int base) {
    v8bf a = *(const v8bf*)(row + base);
    v8bf b = *(const v8bf*)(row + base + 8);
    v16bf r;
#pragma unroll
    for (int i = 0; i < 8; ++i) { r[i] = a[i]; r[8 + i] = b[i]; }
    return r;
}
// B 32x16: lane holds column N=lane&15, 16 consecutive K at k0+(lane>>4)*16.
static __device__ __forceinline__ v16bf ldB_f32(const float* __restrict__ row, int base) {
    v4f x0 = *(const v4f*)(row + base);
    v4f x1 = *(const v4f*)(row + base + 4);
    v4f x2 = *(const v4f*)(row + base + 8);
    v4f x3 = *(const v4f*)(row + base + 12);
    v16bf r;
    r[0]=(__bf16)x0[0]; r[1]=(__bf16)x0[1]; r[2]=(__bf16)x0[2]; r[3]=(__bf16)x0[3];
    r[4]=(__bf16)x1[0]; r[5]=(__bf16)x1[1]; r[6]=(__bf16)x1[2]; r[7]=(__bf16)x1[3];
    r[8]=(__bf16)x2[0]; r[9]=(__bf16)x2[1]; r[10]=(__bf16)x2[2]; r[11]=(__bf16)x2[3];
    r[12]=(__bf16)x3[0]; r[13]=(__bf16)x3[1]; r[14]=(__bf16)x3[2]; r[15]=(__bf16)x3[3];
    return r;
}
static __device__ __forceinline__ v16bf ldB_bf16(const __bf16* __restrict__ row, int base) {
    v8bf a = *(const v8bf*)(row + base);
    v8bf b = *(const v8bf*)(row + base + 8);
    v16bf r;
#pragma unroll
    for (int i = 0; i < 8; ++i) { r[i] = a[i]; r[8 + i] = b[i]; }
    return r;
}
// B fragment from an LDS slab row (16 consecutive bf16, 16B-aligned)
static __device__ __forceinline__ v16bf ldB_lds(const __bf16* p) {
    v8bf a = *(const v8bf*)p;
    v8bf b = *(const v8bf*)(p + 8);
    v16bf r;
#pragma unroll
    for (int i = 0; i < 8; ++i) { r[i] = a[i]; r[8 + i] = b[i]; }
    return r;
}

#define WMMA_BF16(a, b, c) \
    __builtin_amdgcn_wmma_f32_16x16x32_bf16(false, (a), false, (b), (short)0, (c), false, false)

// ---- Stage 1: WqT_bf16[e][j] = bf16(Wq[j][e]) ------------------------------
__global__ void wq_transpose_kernel(const float* __restrict__ Wq, __bf16* __restrict__ WqT) {
    __shared__ float tile[32][33];
    int tx = threadIdx.x, ty = threadIdx.y;          // 32 x 8
    int et = blockIdx.x, jt = blockIdx.y;
#pragma unroll
    for (int i = 0; i < 4; ++i)
        tile[ty * 4 + i][tx] = Wq[(size_t)(jt * 32 + ty * 4 + i) * EE + et * 32 + tx];
    __syncthreads();
#pragma unroll
    for (int i = 0; i < 4; ++i)
        WqT[(size_t)(et * 32 + ty * 4 + i) * EE + jt * 32 + tx] = (__bf16)tile[tx][ty * 4 + i];
}

// ---- Stage 2: K = gather(X,hub) @ Wk^T + bk  (one wave per 16x16 tile) -----
__global__ void k_proj_kernel(const float* __restrict__ X, const int* __restrict__ hub,
                              const float* __restrict__ Wk, const float* __restrict__ bk,
                              float* __restrict__ Kf32, __bf16* __restrict__ Kb) {
    int lane = threadIdx.x & 31;
    int lm = lane & 15, hs = lane >> 4;
    int n0 = blockIdx.x * 16;                        // e tile
    int m0 = blockIdx.y * 16;                        // h tile
    const float* arow = X + (size_t)hub[m0 + lm] * EE;
    const float* brow = Wk + (size_t)(n0 + lm) * EE;
    v8f acc = {};
    for (int k0 = 0; k0 < EE; k0 += 32) {
        v16bf a = ldA_f32(arow, k0 + hs * 16);
        v16bf b = ldB_f32(brow, k0 + hs * 16);
        acc = WMMA_BF16(a, b, acc);
    }
    int e = n0 + lm;
    float bke = bk[e];
#pragma unroll
    for (int v = 0; v < 8; ++v) {
        int h = m0 + hs * 8 + v;
        float val = acc[v] + bke;
        Kf32[(size_t)h * EE + e] = val;
        Kb[(size_t)h * EE + e]   = (__bf16)val;
    }
}

// ---- Stage 3: c[h] = bq . K[h,:] -------------------------------------------
__global__ void cvec_kernel(const float* __restrict__ bq, const float* __restrict__ Kf32,
                            float* __restrict__ cvec) {
    __shared__ float red[256];
    int h = blockIdx.x, t = threadIdx.x;
    float s = 0.f;
    for (int j = t; j < EE; j += 256) s += bq[j] * Kf32[(size_t)h * EE + j];
    red[t] = s; __syncthreads();
    for (int w = 128; w > 0; w >>= 1) { if (t < w) red[t] += red[t + w]; __syncthreads(); }
    if (t == 0) cvec[h] = red[0];
}

// ---- Stage 4: K2 = K @ Wq  (via WqT rows; one wave per 16x16 tile) ---------
__global__ void k2_kernel(const __bf16* __restrict__ Kb, const __bf16* __restrict__ WqT,
                          __bf16* __restrict__ K2) {
    int lane = threadIdx.x & 31;
    int lm = lane & 15, hs = lane >> 4;
    int n0 = blockIdx.x * 16;                        // e tile
    int m0 = blockIdx.y * 16;                        // h tile
    const __bf16* arow = Kb  + (size_t)(m0 + lm) * EE;
    const __bf16* brow = WqT + (size_t)(n0 + lm) * EE;
    v8f acc = {};
    for (int k0 = 0; k0 < EE; k0 += 32) {
        v16bf a = ldA_bf16(arow, k0 + hs * 16);
        v16bf b = ldB_bf16(brow, k0 + hs * 16);
        acc = WMMA_BF16(a, b, acc);
    }
    int e = n0 + lm;
#pragma unroll
    for (int v = 0; v < 8; ++v)
        K2[(size_t)(m0 + hs * 8 + v) * EE + e] = (__bf16)acc[v];
}

// ---- Stage 5: fused scores + argmax ----------------------------------------
// 4 waves/block, 16 rows/wave. K2 slab (256 hubs x 32 k) staged per block in
// LDS via async global->LDS copies, double-buffered under the WMMAs.
#define SLAB_PAD 40   // 80B row stride: conflict-free ds_load_b128 across 16 lanes

__global__ void scores_argmax_kernel(const float* __restrict__ X, const int* __restrict__ hub,
                                     const __bf16* __restrict__ K2, const float* __restrict__ cvec,
                                     float* __restrict__ out) {
    __shared__ __bf16 slab[2][HH][SLAB_PAD];         // 2 x 20KB

    int t = threadIdx.x;                              // 0..127
    int wave = t >> 5, lane = t & 31;
    int lm = lane & 15, hs = lane >> 4;
    size_t row_base = ((size_t)blockIdx.x * 4 + wave) * 16;
    const float* arow = X + (row_base + lm) * EE;

    // each thread stages 2 hub rows (64B each) per k-step, 16B chunks
    const __bf16* g0 = K2 + (size_t)(2 * t) * EE;
    const __bf16* g1 = K2 + (size_t)(2 * t + 1) * EE;

#define ISSUE_SLAB(buf, k0)                                                      \
    do {                                                                         \
        __bf16* l0 = &slab[buf][2 * t][0];                                       \
        __bf16* l1 = &slab[buf][2 * t + 1][0];                                   \
        _Pragma("unroll")                                                        \
        for (int c = 0; c < 4; ++c) {                                            \
            ASYNC_CP16(g0 + (k0) + c * 8, l0 + c * 8);                           \
            ASYNC_CP16(g1 + (k0) + c * 8, l1 + c * 8);                           \
        }                                                                        \
    } while (0)

    v8f acc[16] = {};
    ISSUE_SLAB(0, 0);
    for (int it = 0; it < EE / 32; ++it) {
        int k0 = it * 32;
        ASYNC_WAIT();                 // my slab writes landed in LDS
        __syncthreads();              // everyone's landed; old buffer free
        if (it + 1 < EE / 32) ISSUE_SLAB((it + 1) & 1, k0 + 32);

        __builtin_prefetch(arow + k0 + 256, 0, 0);   // global_prefetch_b8 on X stream
        v16bf a = ldA_f32(arow, k0 + hs * 16);
        const __bf16* sb = &slab[it & 1][0][0];
#pragma unroll
        for (int f = 0; f < 16; ++f) {
            v16bf b = ldB_lds(sb + (f * 16 + lm) * SLAB_PAD + hs * 16);
            acc[f] = WMMA_BF16(a, b, acc[f]);
        }
    }
#undef ISSUE_SLAB

    float cv[16];
#pragma unroll
    for (int f = 0; f < 16; ++f) cv[f] = cvec[f * 16 + lm];

#pragma unroll
    for (int v = 0; v < 8; ++v) {
        float bv = -3.402823466e38f; int bh = 0;
#pragma unroll
        for (int f = 0; f < 16; ++f) {
            float val = acc[f][v] + cv[f];
            int h = f * 16 + lm;
            if (val > bv || (val == bv && h < bh)) { bv = val; bh = h; }
        }
        // butterfly max-reduce inside each 16-lane half (wave32)
#pragma unroll
        for (int off = 8; off >= 1; off >>= 1) {
            float ov = __shfl_xor(bv, off, 32);
            int   oh = __shfl_xor(bh, off, 32);
            if (ov > bv || (ov == bv && oh < bh)) { bv = ov; bh = oh; }
        }
        if (lm == 0) {
            size_t node = row_base + hs * 8 + v;
            int lo = 0, hi = HH - 1, found = -1;
            while (lo <= hi) {
                int mid = (lo + hi) >> 1;
                int hvn = hub[mid];
                if (hvn == (int)node) { found = mid; break; }
                else if (hvn < (int)node) lo = mid + 1;
                else hi = mid - 1;
            }
            int assign = (found >= 0) ? (int)node : hub[bh];
            out[node] = (float)assign;
        }
    }
}

extern "C" void kernel_launch(void* const* d_in, const int* in_sizes, int n_in,
                              void* d_out, int out_size, void* d_ws, size_t ws_size,
                              hipStream_t stream) {
    (void)in_sizes; (void)n_in; (void)out_size; (void)ws_size;
    const float* X   = (const float*)d_in[0];
    const int*   hub = (const int*)d_in[1];
    const float* Wq  = (const float*)d_in[2];
    const float* bq  = (const float*)d_in[3];
    const float* Wk  = (const float*)d_in[4];
    const float* bk  = (const float*)d_in[5];
    float* out = (float*)d_out;

    char* ws = (char*)d_ws;
    size_t off = 0;
    __bf16* WqT  = (__bf16*)(ws + off); off += (size_t)EE * EE * 2;   // 32 MB
    float*  Kf32 = (float*)(ws + off);  off += (size_t)HH * EE * 4;   // 4 MB
    __bf16* Kb   = (__bf16*)(ws + off); off += (size_t)HH * EE * 2;   // 2 MB
    __bf16* K2   = (__bf16*)(ws + off); off += (size_t)HH * EE * 2;   // 2 MB
    float*  cvec = (float*)(ws + off);  off += (size_t)HH * 4;

    wq_transpose_kernel<<<dim3(EE / 32, EE / 32), dim3(32, 8), 0, stream>>>(Wq, WqT);
    k_proj_kernel<<<dim3(EE / 16, HH / 16), 32, 0, stream>>>(X, hub, Wk, bk, Kf32, Kb);
    cvec_kernel<<<HH, 256, 0, stream>>>(bq, Kf32, cvec);
    k2_kernel<<<dim3(EE / 16, HH / 16), 32, 0, stream>>>(Kb, WqT, K2);
    // 16 rows per wave, 4 waves per block -> 64 rows/block, 256 blocks
    scores_argmax_kernel<<<NN / 64, 128, 0, stream>>>(X, hub, K2, cvec, out);
}
